// CrossAttention_9732395893238
// MI455X (gfx1250) — matrix operand order repaired and checked by reading, not compiled, not run on hardware
//
#include <hip/hip_runtime.h>

// ---------- types ----------
typedef __attribute__((ext_vector_type(16))) __bf16 v16bf;
typedef __attribute__((ext_vector_type(8)))  float  v8f;
typedef __attribute__((ext_vector_type(4)))  unsigned int u32x4;
typedef __attribute__((ext_vector_type(2)))  unsigned int u32x2;
typedef __attribute__((ext_vector_type(4)))  float  f32x4;
typedef __attribute__((ext_vector_type(4)))  int    v4i;

union FragB { v16bf v; u32x4 q[2]; };
union FragC { v8f v; float f[8]; };

// ---------- async Global->LDS staging (CDNA5 ASYNCcnt path) ----------
#if defined(__has_builtin)
#if __has_builtin(__builtin_amdgcn_global_load_async_to_lds_b128)
#define HAVE_ASYNC_LDS 1
#endif
#endif
#ifndef HAVE_ASYNC_LDS
#define HAVE_ASYNC_LDS 0
#endif

#if HAVE_ASYNC_LDS
typedef __attribute__((address_space(1))) v4i* g_v4i;  // global (prints as __device__)
typedef __attribute__((address_space(3))) v4i* l_v4i;  // LDS    (prints as __shared__)
#endif

// copy 16 bytes global -> LDS (async engine when available)
__device__ __forceinline__ void async_copy16(const unsigned short* g,
                                             unsigned short* l) {
#if HAVE_ASYNC_LDS
    __builtin_amdgcn_global_load_async_to_lds_b128(
        (g_v4i)(unsigned short*)g, (l_v4i)l, 0, 0);
#else
    *(u32x4*)l = *(const u32x4*)g;
#endif
}

__device__ __forceinline__ void async_join() {
#if HAVE_ASYNC_LDS
    asm volatile("s_wait_asynccnt 0x0" ::: "memory");
#endif
}

// float -> bf16 (round to nearest even)
__device__ __forceinline__ unsigned short f2bf(float x) {
    unsigned u = __builtin_bit_cast(unsigned, x);
    u += 0x7FFFu + ((u >> 16) & 1u);
    return (unsigned short)(u >> 16);
}

__device__ __forceinline__ v8f wmma_bf16(v16bf a, v16bf b, v8f c) {
    return __builtin_amdgcn_wmma_f32_16x16x32_bf16(false, a, false, b, (short)0, c,
                                                   false, false);
}

// A-fragment (16x32 bf16) from a [row][k] tile, row-stride ld (elements).
// lane: row = lane&15 ; elems 0..7 = K[koff..koff+7], 8..15 = K[16+koff..], koff=8*(lane>=16)
__device__ __forceinline__ v16bf ldA(const unsigned short* p, int ld) {
    int l = threadIdx.x & 31;
    const unsigned short* r = p + (l & 15) * ld + ((l >> 4) << 3);
    FragB f;
    f.q[0] = *(const u32x4*)r;
    f.q[1] = *(const u32x4*)(r + 16);
    return f.v;
}

// B-fragment (32x16 bf16) from a [col][k] tile, col-stride ld (elements).
// lane: col = lane&15 ; elems 0..7 = K[koff..koff+7], 8..15 = K[16+koff..], koff=8*(lane>=16)
__device__ __forceinline__ v16bf ldB(const unsigned short* p, int ld) {
    int l = threadIdx.x & 31;
    const unsigned short* r = p + (l & 15) * ld + ((l >> 4) << 3);
    FragB f;
    f.q[0] = *(const u32x4*)r;
    f.q[1] = *(const u32x4*)(r + 16);
    return f.v;
}

__device__ __forceinline__ u32x4 packC(const FragC& c) {
    u32x4 u;
#pragma unroll
    for (int k = 0; k < 4; ++k)
        u[k] = (unsigned)f2bf(c.f[2 * k]) | ((unsigned)f2bf(c.f[2 * k + 1]) << 16);
    return u;
}

// =====================================================================
// proj_kernel: Y = X[rows x 1024] @ W[1024 x ldw]  (fp32 -> bf16)
//   cols < vColBase : store row-major into outRM (ld 1024)  [Q / K]
//   cols >= vColBase: store transposed into outTR as [b*1024+dg][2048] [V^T]
// =====================================================================
__global__ __launch_bounds__(256) void proj_kernel(
    const float* __restrict__ X, const float* __restrict__ W, int ldw,
    unsigned short* __restrict__ outRM, unsigned short* __restrict__ outTR,
    int vColBase)
{
    __shared__ __align__(16) unsigned short lx[128 * 32];
    __shared__ __align__(16) unsigned short lw[128 * 32];
    const int tid = threadIdx.x, wid = tid >> 5, lane = tid & 31;
    const int c0 = blockIdx.x * 128, r0 = blockIdx.y * 128;
    const bool rm = (c0 < vColBase);
    const int wc = (wid & 3) * 32;   // wave col block (W cols)
    const int wr = (wid >> 2) * 64;  // wave row block (X rows)

    FragC acc[2][4];
#pragma unroll
    for (int j = 0; j < 2; ++j)
#pragma unroll
        for (int i = 0; i < 4; ++i)
#pragma unroll
            for (int k = 0; k < 8; ++k) acc[j][i].f[k] = 0.f;

    for (int kt = 0; kt < 1024; kt += 32) {
        // stage X tile (128 rows x 32 k), fp32 -> bf16 (VALU convert path)
#pragma unroll
        for (int it = 0; it < 4; ++it) {
            int v = tid + it * 256;            // 1024 float4 units
            int row = v >> 3, cc = (v & 7) * 4;
            f32x4 f = *(const f32x4*)&X[(size_t)(r0 + row) * 1024 + kt + cc];
            u32x2 u;
            u[0] = (unsigned)f2bf(f[0]) | ((unsigned)f2bf(f[1]) << 16);
            u[1] = (unsigned)f2bf(f[2]) | ((unsigned)f2bf(f[3]) << 16);
            *(u32x2*)&lx[row * 32 + cc] = u;
        }
        // stage W tile transposed: lw[n][k] (n=128 cols, k=32)
#pragma unroll
        for (int it = 0; it < 4; ++it) {
            int v = tid + it * 256;            // 1024 float4 units
            int k = v >> 5, cc = (v & 31) * 4;
            f32x4 f = *(const f32x4*)&W[(size_t)(kt + k) * ldw + c0 + cc];
            lw[(cc + 0) * 32 + k] = f2bf(f[0]);
            lw[(cc + 1) * 32 + k] = f2bf(f[1]);
            lw[(cc + 2) * 32 + k] = f2bf(f[2]);
            lw[(cc + 3) * 32 + k] = f2bf(f[3]);
        }
        __syncthreads();
        if (rm) {
            // C^T = W^T * X^T : A = W^T rows (w cols), B = X rows
            v16bf wf[2], xf[4];
#pragma unroll
            for (int j = 0; j < 2; ++j) wf[j] = ldA(&lw[(wc + j * 16) * 32], 32);
#pragma unroll
            for (int i = 0; i < 4; ++i) xf[i] = ldB(&lx[(wr + i * 16) * 32], 32);
#pragma unroll
            for (int j = 0; j < 2; ++j)
#pragma unroll
                for (int i = 0; i < 4; ++i)
                    acc[j][i].v = wmma_bf16(wf[j], xf[i], acc[j][i].v);
        } else {
            // C = X * W : A = X rows, B = W^T rows (w cols)
            v16bf xf[4], wf[2];
#pragma unroll
            for (int i = 0; i < 4; ++i) xf[i] = ldA(&lx[(wr + i * 16) * 32], 32);
#pragma unroll
            for (int j = 0; j < 2; ++j) wf[j] = ldB(&lw[(wc + j * 16) * 32], 32);
#pragma unroll
            for (int j = 0; j < 2; ++j)
#pragma unroll
                for (int i = 0; i < 4; ++i)
                    acc[j][i].v = wmma_bf16(xf[i], wf[j], acc[j][i].v);
        }
        __syncthreads();
    }

    if (rm) {
        // C^T tile [M=wcol][N=xrow]: lane holds xrow=lane&15, 8 contiguous wcols
#pragma unroll
        for (int j = 0; j < 2; ++j)
#pragma unroll
            for (int i = 0; i < 4; ++i) {
                int gr = r0 + wr + i * 16 + (lane & 15);
                int gc = c0 + wc + j * 16 + ((lane >> 4) << 3);
                *(u32x4*)&outRM[(size_t)gr * 1024 + gc] = packC(acc[j][i]);
            }
    } else {
        // C tile [M=xrow][N=wcol]: lane holds wcol=lane&15, 8 contiguous xrows (m)
#pragma unroll
        for (int j = 0; j < 2; ++j)
#pragma unroll
            for (int i = 0; i < 4; ++i) {
                int dg = (c0 - vColBase) + wc + j * 16 + (lane & 15);
                int grow = r0 + wr + i * 16 + ((lane >> 4) << 3);
                int b = grow >> 11, m = grow & 2047;
                *(u32x4*)&outTR[((size_t)(b * 1024 + dg)) * 2048 + m] =
                    packC(acc[j][i]);
            }
    }
}

// =====================================================================
// attn_kernel: flash attention per (b, h, 128-row n block); 8 waves x 16 rows
//   Q,K row-major bf16 [b*2048+n][1024]; Vt bf16 [b*1024 + h*64 + d][2048]
//   K / V^T tiles staged with async Global->LDS (ASYNCcnt) when available
// =====================================================================
__global__ __launch_bounds__(256) void attn_kernel(
    const unsigned short* __restrict__ Q, const unsigned short* __restrict__ K,
    const unsigned short* __restrict__ Vt,
    const float* __restrict__ xmask, const float* __restrict__ cmask,
    unsigned short* __restrict__ Aout)
{
    __shared__ __align__(16) unsigned short lk[64 * 64];
    __shared__ __align__(16) unsigned short lv[64 * 64];
    __shared__ __align__(16) unsigned short lp[8][16 * 32];
    const int tid = threadIdx.x, wid = tid >> 5, lane = tid & 31;
    const int b = blockIdx.z, h = blockIdx.y;
    const int n0 = blockIdx.x * 128 + wid * 16;
    const int rbase = (lane >> 4) << 3;

    // Q A-fragments for d = 0..31 and 32..63, held in VGPRs for the whole kernel
    v16bf qf[2];
    {
        const unsigned short* qb = Q + ((size_t)(b * 2048 + n0 + (lane & 15))) * 1024 +
                                   h * 64 + ((lane >> 4) << 3);
#pragma unroll
        for (int c = 0; c < 2; ++c) {
            FragB f;
            f.q[0] = *(const u32x4*)(qb + c * 32);
            f.q[1] = *(const u32x4*)(qb + c * 32 + 16);
            qf[c] = f.v;
        }
    }

    FragC o[4];
#pragma unroll
    for (int t = 0; t < 4; ++t)
#pragma unroll
        for (int k = 0; k < 8; ++k) o[t].f[k] = 0.f;
    float rmx[8], rsm[8], kvmax = 0.f;
#pragma unroll
    for (int r = 0; r < 8; ++r) { rmx[r] = -1e30f; rsm[r] = 0.f; }

    const unsigned short* kb = K + ((size_t)(b * 2048)) * 1024 + h * 64;
    const unsigned short* vb = Vt + ((size_t)(b * 1024 + h * 64)) * 2048;
    const float* cmrow = cmask + b * 2048;

    for (int m0 = 0; m0 < 2048; m0 += 64) {
        // cooperative stage: K tile [64 m][64 d], V^T tile [64 d][64 m]
#pragma unroll
        for (int it = 0; it < 2; ++it) {
            int v = tid + it * 256;           // 512 x 16B units
            int row = v >> 3, cc = (v & 7) * 8;
            async_copy16(&kb[(size_t)(m0 + row) * 1024 + cc], &lk[row * 64 + cc]);
            async_copy16(&vb[(size_t)row * 2048 + m0 + cc],   &lv[row * 64 + cc]);
        }
        async_join();
        __syncthreads();
#pragma unroll
        for (int s = 0; s < 64; s += 32) {
            // S = Q K^T for two 16-col subtiles (d split in 2 x K=32)
            FragC sa, sb;
            {
                v8f z = {0.f, 0.f, 0.f, 0.f, 0.f, 0.f, 0.f, 0.f};
                v8f t0 = wmma_bf16(qf[0], ldB(&lk[s * 64 + 0], 64), z);
                sa.v    = wmma_bf16(qf[1], ldB(&lk[s * 64 + 32], 64), t0);
                v8f t1 = wmma_bf16(qf[0], ldB(&lk[(s + 16) * 64 + 0], 64), z);
                sb.v    = wmma_bf16(qf[1], ldB(&lk[(s + 16) * 64 + 32], 64), t1);
            }
            float cma = cmrow[m0 + s + (lane & 15)];
            float cmb = cmrow[m0 + s + 16 + (lane & 15)];
            kvmax = fmaxf(kvmax, fmaxf(cma, cmb));
            const float NEG = -1e30f, SC = 0.125f;  // DIM_HEAD^-0.5
            float pa[8], pb[8];
#pragma unroll
            for (int r = 0; r < 8; ++r) {
                float a  = (cma > 0.f) ? sa.f[r] * SC : NEG;
                float bb = (cmb > 0.f) ? sb.f[r] * SC : NEG;
                float mx = fmaxf(a, bb);
#pragma unroll
                for (int d = 1; d < 16; d <<= 1) mx = fmaxf(mx, __shfl_xor(mx, d, 32));
                float mnew  = fmaxf(rmx[r], mx);
                float alpha = __expf(rmx[r] - mnew);
                float ea = __expf(a - mnew), eb = __expf(bb - mnew);
                pa[r] = ea; pb[r] = eb;
                float ps = ea + eb;
#pragma unroll
                for (int d = 1; d < 16; d <<= 1) ps += __shfl_xor(ps, d, 32);
                rsm[r] = rsm[r] * alpha + ps;
                rmx[r] = mnew;
#pragma unroll
                for (int t = 0; t < 4; ++t) o[t].f[r] *= alpha;
            }
            // transpose P (C layout -> A-fragment layout) through wave-private LDS
            unsigned short* lpw = lp[wid];
            const int col = lane & 15;
#pragma unroll
            for (int r = 0; r < 8; ++r) {
                lpw[(rbase + r) * 32 + col]      = f2bf(pa[r]);
                lpw[(rbase + r) * 32 + col + 16] = f2bf(pb[r]);
            }
            v16bf pf = ldA(lpw, 32);
#pragma unroll
            for (int t = 0; t < 4; ++t)
                o[t].v = wmma_bf16(pf, ldB(&lv[(t * 16) * 64 + s], 64), o[t].v);
        }
        __syncthreads();
    }

    // any valid key? (each 16-lane group collectively saw every m column)
#pragma unroll
    for (int d = 1; d < 16; d <<= 1) kvmax = fmaxf(kvmax, __shfl_xor(kvmax, d, 32));
    float inv[8];
#pragma unroll
    for (int r = 0; r < 8; ++r) {
        float xm = xmask[b * 2048 + n0 + rbase + r];
        inv[r] = (xm > 0.f && kvmax > 0.f && rsm[r] > 0.f) ? 1.f / rsm[r] : 0.f;
    }
    const int col = lane & 15;
#pragma unroll
    for (int t = 0; t < 4; ++t)
#pragma unroll
        for (int r = 0; r < 8; ++r)
            Aout[(size_t)(b * 2048 + n0 + rbase + r) * 1024 + h * 64 + t * 16 + col] =
                f2bf(o[t].f[r] * inv[r]);
}

// =====================================================================
// out_gemm_kernel: out = A(bf16)[8192x1024] @ Wo(fp32)[1024x1024] + bo  (fp32 out)
//   bf16 A tile staged with async Global->LDS when available
// =====================================================================
__global__ __launch_bounds__(256) void out_gemm_kernel(
    const unsigned short* __restrict__ A, const float* __restrict__ W,
    const float* __restrict__ bias, float* __restrict__ out)
{
    __shared__ __align__(16) unsigned short lx[128 * 32];
    __shared__ __align__(16) unsigned short lw[128 * 32];
    const int tid = threadIdx.x, wid = tid >> 5, lane = tid & 31;
    const int c0 = blockIdx.x * 128, r0 = blockIdx.y * 128;
    const int wc = (wid & 3) * 32, wr = (wid >> 2) * 64;

    FragC acc[2][4];
#pragma unroll
    for (int j = 0; j < 2; ++j)
#pragma unroll
        for (int i = 0; i < 4; ++i)
#pragma unroll
            for (int k = 0; k < 8; ++k) acc[j][i].f[k] = 0.f;

    for (int kt = 0; kt < 1024; kt += 32) {
#pragma unroll
        for (int it = 0; it < 2; ++it) {   // bf16 A tile: async 16B copies
            int v = tid + it * 256;        // 512 x 16B units
            int row = v >> 2, cc = (v & 3) * 8;
            async_copy16(&A[(size_t)(r0 + row) * 1024 + kt + cc],
                         &lx[row * 32 + cc]);
        }
#pragma unroll
        for (int it = 0; it < 4; ++it) {   // Wo fp32 -> bf16, transposed
            int v = tid + it * 256;
            int k = v >> 5, cc = (v & 31) * 4;
            f32x4 f = *(const f32x4*)&W[(size_t)(kt + k) * 1024 + c0 + cc];
            lw[(cc + 0) * 32 + k] = f2bf(f[0]);
            lw[(cc + 1) * 32 + k] = f2bf(f[1]);
            lw[(cc + 2) * 32 + k] = f2bf(f[2]);
            lw[(cc + 3) * 32 + k] = f2bf(f[3]);
        }
        async_join();
        __syncthreads();
        v16bf wf[2], xf[4];
#pragma unroll
        for (int j = 0; j < 2; ++j) wf[j] = ldA(&lw[(wc + j * 16) * 32], 32);
#pragma unroll
        for (int i = 0; i < 4; ++i) xf[i] = ldB(&lx[(wr + i * 16) * 32], 32);
#pragma unroll
        for (int j = 0; j < 2; ++j)
#pragma unroll
            for (int i = 0; i < 4; ++i)
                acc[j][i].v = wmma_bf16(wf[j], xf[i], acc[j][i].v);
        __syncthreads();
    }

#pragma unroll
    for (int j = 0; j < 2; ++j)
#pragma unroll
        for (int i = 0; i < 4; ++i) {
            int gr = r0 + wr + i * 16 + (lane & 15);
            int gc = c0 + wc + j * 16 + ((lane >> 4) << 3);
            f32x4 b0 = *(const f32x4*)&bias[gc];
            f32x4 b1 = *(const f32x4*)&bias[gc + 4];
            f32x4 o0, o1;
#pragma unroll
            for (int k = 0; k < 4; ++k) {
                o0[k] = acc[j][i].f[k] + b0[k];
                o1[k] = acc[j][i].f[4 + k] + b1[k];
            }
            *(f32x4*)&out[(size_t)gr * 1024 + gc] = o0;
            *(f32x4*)&out[(size_t)gr * 1024 + gc + 4] = o1;
        }
}

// =====================================================================
extern "C" void kernel_launch(void* const* d_in, const int* in_sizes, int n_in,
                              void* d_out, int out_size, void* d_ws, size_t ws_size,
                              hipStream_t stream)
{
    (void)in_sizes; (void)n_in; (void)out_size; (void)ws_size;
    const float* x   = (const float*)d_in[0];   // (4,2048,1024)
    const float* ctx = (const float*)d_in[1];   // (4,2048,1024)
    const float* xm  = (const float*)d_in[2];   // (4,2048)
    const float* cm  = (const float*)d_in[3];   // (4,2048)
    const float* Wq  = (const float*)d_in[4];   // (1024,1024)
    const float* Wkv = (const float*)d_in[5];   // (1024,2048)
    const float* Wo  = (const float*)d_in[6];   // (1024,1024)
    const float* bo  = (const float*)d_in[7];   // (1024)
    float* out = (float*)d_out;                 // (4,2048,1024)

    const size_t SZ = (size_t)8192 * 1024;      // elements per bf16 buffer
    unsigned short* Qb = (unsigned short*)d_ws; // 16 MB
    unsigned short* Kb = Qb + SZ;               // 16 MB
    unsigned short* Vt = Kb + SZ;               // 16 MB  [b*1024 + h*64 + d][2048]
    unsigned short* Ao = Vt + SZ;               // 16 MB

    // Q = x @ Wq  (all row-major)
    proj_kernel<<<dim3(8, 64), 256, 0, stream>>>(x, Wq, 1024, Qb, Vt, 1024);
    // [K|V] = context @ Wkv ; K row-major, V stored transposed
    proj_kernel<<<dim3(16, 64), 256, 0, stream>>>(ctx, Wkv, 2048, Kb, Vt, 1024);
    // flash attention -> attn_out (bf16 row-major)
    attn_kernel<<<dim3(16, 16, 4), 256, 0, stream>>>(Qb, Kb, Vt, xm, cm, Ao);
    // out = attn_out @ Wo + bo
    out_gemm_kernel<<<dim3(8, 64), 256, 0, stream>>>(Ao, Wo, bo, out);
}
